// LatentMoE_84129819394135
// MI455X (gfx1250) — matrix operand order repaired and checked by reading, not compiled, not run on hardware
//
#include <hip/hip_runtime.h>
#include <hip/hip_bf16.h>

typedef __bf16 bf16_t;
typedef bf16_t v16bf __attribute__((ext_vector_type(16)));
typedef bf16_t v8bf  __attribute__((ext_vector_type(8)));
typedef float  v8f   __attribute__((ext_vector_type(8)));

#define TOKENS 8192
#define EMB 2048
#define LAT 512
#define HID 1024
#define NEXP 16
#define TOPK 8
#define RSCALE 2.5f

#define TMB 256   // block tile M
#define TNB 128   // block tile N
#define TK  32    // K step
#define LDSS 40   // LDS row stride in bf16 units (80 B -> 16B-aligned b128 accesses)

enum { EPI_F32 = 0, EPI_BF16, EPI_SQRELU, EPI_MUL, EPI_ADDF32, EPI_ADDRS };

// ---- CDNA5 async global->LDS copy (ASYNCcnt-tracked), 16 bytes per lane ----
__device__ __forceinline__ void async_copy16B(unsigned lds_addr, const bf16_t* gptr) {
  asm volatile("global_load_async_to_lds_b128 %0, %1, off"
               :: "v"(lds_addr), "v"((unsigned long long)(uintptr_t)gptr)
               : "memory");
}
__device__ __forceinline__ void wait_async0() {
  asm volatile("s_wait_asynccnt 0" ::: "memory");
}

// -------- tiled bf16 WMMA GEMM: C[M,N] (epilogue) = A[M,K] @ B[N,K]^T --------
// Block: 256 threads (8 waves). Tile 256x128, K-step 32, double-buffered async LDS.
// Waves in 4(M)x2(N) grid -> 64x64 per wave -> 16 v_wmma_f32_16x16x32_bf16 per step.
template<int EPI>
__global__ __launch_bounds__(256) void gemm_bf16_wmma(
    const bf16_t* __restrict__ A, const bf16_t* __restrict__ B,
    void* __restrict__ Cv, const bf16_t* __restrict__ mulbuf,
    const float* __restrict__ rowscale, int rs_stride,
    int M, int N, int K)
{
  __shared__ __align__(16) bf16_t As[2][TMB * LDSS];
  __shared__ __align__(16) bf16_t Bs[2][TNB * LDSS];

  const int tid  = threadIdx.x;
  const int lane = tid & 31, wave = tid >> 5;
  const int l16  = lane & 15, half = lane >> 4;
  const int wm = (wave >> 1) * 64;   // 4 wave-rows x 64 M
  const int wn = (wave & 1) * 64;    // 2 wave-cols x 64 N
  const long m0 = (long)blockIdx.y * TMB;
  const long n0 = (long)blockIdx.x * TNB;

  const unsigned aLds[2] = { (unsigned)(uintptr_t)&As[0][0], (unsigned)(uintptr_t)&As[1][0] };
  const unsigned bLds[2] = { (unsigned)(uintptr_t)&Bs[0][0], (unsigned)(uintptr_t)&Bs[1][0] };

  v8f acc[4][4];
#pragma unroll
  for (int i = 0; i < 4; i++)
#pragma unroll
    for (int j = 0; j < 4; j++)
#pragma unroll
      for (int r = 0; r < 8; r++) acc[i][j][r] = 0.f;

  // stage one K-step of A (256x32) and B (128x32) via async b128 copies
  auto stage = [&](int buf, long k0) {
    // A: 1024 chunks of 8 bf16 (16B); 4 per thread
#pragma unroll
    for (int q = 0; q < 4; q++) {
      int c = tid + 256 * q;
      int row = c >> 2, part = c & 3;
      async_copy16B(aLds[buf] + (unsigned)(row * LDSS + part * 8) * 2u,
                    A + (m0 + row) * (long)K + k0 + part * 8);
    }
    // B: 512 chunks; 2 per thread
#pragma unroll
    for (int q = 0; q < 2; q++) {
      int c = tid + 256 * q;
      int row = c >> 2, part = c & 3;
      async_copy16B(bLds[buf] + (unsigned)(row * LDSS + part * 8) * 2u,
                    B + (n0 + row) * (long)K + k0 + part * 8);
    }
  };

  const int nk = K / TK;
  stage(0, 0);
  for (int kt = 0; kt < nk; ++kt) {
    wait_async0();          // this wave's copies into buf cur are complete
    __syncthreads();        // all waves' copies complete; prev reads done
    const int cur = kt & 1;
    if (kt + 1 < nk) stage(cur ^ 1, (long)(kt + 1) * TK);

    const bf16_t* Asp = &As[cur][0];
    const bf16_t* Bsp = &Bs[cur][0];

    // A fragments (16x32): lanes 0-15 K=0..7|16..23, lanes 16-31 K=8..15|24..31
    v16bf af[4];
#pragma unroll
    for (int i = 0; i < 4; i++) {
      const bf16_t* ap = Asp + (wm + i * 16 + l16) * LDSS;
      v8bf lo = *(const v8bf*)(ap + half * 8);
      v8bf hi = *(const v8bf*)(ap + 16 + half * 8);
      af[i] = __builtin_shufflevector(lo, hi, 0,1,2,3,4,5,6,7,8,9,10,11,12,13,14,15);
    }
    // B fragments (32x16): lanes 0-15 K=0..7|16..23, lanes 16-31 K=8..15|24..31
    v16bf bfr[4];
#pragma unroll
    for (int j = 0; j < 4; j++) {
      const bf16_t* bp = Bsp + (wn + j * 16 + l16) * LDSS + half * 8;
      v8bf lo = *(const v8bf*)(bp);
      v8bf hi = *(const v8bf*)(bp + 16);
      bfr[j] = __builtin_shufflevector(lo, hi, 0,1,2,3,4,5,6,7,8,9,10,11,12,13,14,15);
    }
#pragma unroll
    for (int i = 0; i < 4; i++)
#pragma unroll
      for (int j = 0; j < 4; j++)
        acc[i][j] = __builtin_amdgcn_wmma_f32_16x16x32_bf16(
            false, af[i], false, bfr[j], (short)0, acc[i][j], false, false);
    __syncthreads();        // everyone done reading buf cur before it is re-filled
  }

  // epilogue; C layout: VGPR r -> (M = r or r+8 by lane-half, N = lane%16)
#pragma unroll
  for (int i = 0; i < 4; i++) {
#pragma unroll
    for (int j = 0; j < 4; j++) {
#pragma unroll
      for (int r = 0; r < 8; r++) {
        long m = m0 + wm + i * 16 + (half ? r + 8 : r);
        long n = n0 + wn + j * 16 + l16;
        long idx = m * (long)N + n;
        float v = acc[i][j][r];
        if constexpr (EPI == EPI_F32) {
          ((float*)Cv)[idx] = v;
        } else if constexpr (EPI == EPI_BF16) {
          ((bf16_t*)Cv)[idx] = (bf16_t)v;
        } else if constexpr (EPI == EPI_SQRELU) {
          float t = v > 0.f ? v : 0.f;
          ((bf16_t*)Cv)[idx] = (bf16_t)(t * t);
        } else if constexpr (EPI == EPI_MUL) {
          ((bf16_t*)Cv)[idx] = (bf16_t)(v * (float)mulbuf[idx]);
        } else if constexpr (EPI == EPI_ADDF32) {
          ((float*)Cv)[idx] += v;
        } else { // EPI_ADDRS
          ((float*)Cv)[idx] += rowscale[m * (long)rs_stride] * v;
        }
      }
    }
  }
}

// -------- f32 -> bf16 conversion (8 elems/thread) --------
__global__ __launch_bounds__(256) void cvt_bf16_kernel(
    const float* __restrict__ s, bf16_t* __restrict__ d, long n)
{
  long i = (blockIdx.x * (long)blockDim.x + threadIdx.x) * 8;
  if (i >= n) return;
  float4 f0 = *(const float4*)(s + i);
  float4 f1 = *(const float4*)(s + i + 4);
  v8bf o;
  o[0] = (bf16_t)f0.x; o[1] = (bf16_t)f0.y; o[2] = (bf16_t)f0.z; o[3] = (bf16_t)f0.w;
  o[4] = (bf16_t)f1.x; o[5] = (bf16_t)f1.y; o[6] = (bf16_t)f1.z; o[7] = (bf16_t)f1.w;
  *(v8bf*)(d + i) = o;
}

// -------- gate: probas[t,e] = sigmoid(x[t] . Wg[e]), fp32 for exact top-k order ----
__global__ __launch_bounds__(128) void gate_kernel(
    const float* __restrict__ x, const float* __restrict__ Wg, float* __restrict__ probas)
{
  __shared__ float red[NEXP][128];
  const int t = blockIdx.x, tid = threadIdx.x;
  float p[NEXP];
#pragma unroll
  for (int e = 0; e < NEXP; e++) p[e] = 0.f;
  const float* xr = x + (long)t * EMB;
  for (int k = tid; k < EMB; k += 128) {
    float xv = xr[k];
#pragma unroll
    for (int e = 0; e < NEXP; e++) p[e] += xv * Wg[e * EMB + k];
  }
#pragma unroll
  for (int e = 0; e < NEXP; e++) red[e][tid] = p[e];
  __syncthreads();
  for (int s = 64; s > 0; s >>= 1) {
    if (tid < s) {
#pragma unroll
      for (int e = 0; e < NEXP; e++) red[e][tid] += red[e][tid + s];
    }
    __syncthreads();
  }
  if (tid < NEXP) probas[(long)t * NEXP + tid] = 1.f / (1.f + expf(-red[tid][0]));
}

// -------- routing: top-8 of 16, renormalize, scale; dense combine weights ----
__global__ __launch_bounds__(256) void route_kernel(
    const float* __restrict__ probas, const float* __restrict__ biases,
    float* __restrict__ wdense)
{
  int t = blockIdx.x * 256 + threadIdx.x;
  if (t >= TOKENS) return;
  float p[NEXP], w[NEXP];
  bool used[NEXP];
#pragma unroll
  for (int e = 0; e < NEXP; e++) {
    p[e] = probas[(long)t * NEXP + e];
    used[e] = false;
    w[e] = 0.f;
  }
  float psum = 0.f;
  int sel[TOPK];
  for (int k = 0; k < TOPK; k++) {
    int best = 0; float bv = -1e30f;
    for (int e = 0; e < NEXP; e++) {
      float s = p[e] + biases[e];
      if (!used[e] && s > bv) { bv = s; best = e; }
    }
    used[best] = true; sel[k] = best; psum += p[best];
  }
  float inv = RSCALE / psum;
  for (int k = 0; k < TOPK; k++) w[sel[k]] = p[sel[k]] * inv;
#pragma unroll
  for (int e = 0; e < NEXP; e++) wdense[(long)t * NEXP + e] = w[e];
}

__global__ void fill0_kernel(float* __restrict__ p, long n) {
  long i = blockIdx.x * (long)blockDim.x + threadIdx.x;
  if (i < n) p[i] = 0.f;
}

extern "C" void kernel_launch(void* const* d_in, const int* in_sizes, int n_in,
                              void* d_out, int out_size, void* d_ws, size_t ws_size,
                              hipStream_t stream) {
  (void)in_sizes; (void)n_in; (void)out_size; (void)ws_size;
  const float* x      = (const float*)d_in[0];
  const float* sw1    = (const float*)d_in[1];
  const float* swg    = (const float*)d_in[2];
  const float* sw2    = (const float*)d_in[3];
  const float* rw1    = (const float*)d_in[4];
  const float* rwg    = (const float*)d_in[5];
  const float* rw2    = (const float*)d_in[6];
  const float* Wg     = (const float*)d_in[7];
  const float* Wdown  = (const float*)d_in[8];
  const float* Wup    = (const float*)d_in[9];
  const float* biases = (const float*)d_in[10];
  float* out = (float*)d_out;

  char* ws = (char*)d_ws;
  size_t off = 0;
  auto walloc = [&](size_t bytes) {
    void* p = ws + off;
    off += (bytes + 255) & ~(size_t)255;
    return p;
  };
  // bf16 copies of activations + weights (async LDS path needs bf16 in memory)
  bf16_t* xbf    = (bf16_t*)walloc((size_t)TOKENS * EMB * 2);
  bf16_t* sw1b   = (bf16_t*)walloc((size_t)HID * EMB * 2);
  bf16_t* swgb   = (bf16_t*)walloc((size_t)HID * EMB * 2);
  bf16_t* sw2b   = (bf16_t*)walloc((size_t)EMB * HID * 2);
  bf16_t* rw1b   = (bf16_t*)walloc((size_t)NEXP * HID * LAT * 2);
  bf16_t* rwgb   = (bf16_t*)walloc((size_t)NEXP * HID * LAT * 2);
  bf16_t* rw2b   = (bf16_t*)walloc((size_t)NEXP * LAT * HID * 2);
  bf16_t* Wdownb = (bf16_t*)walloc((size_t)LAT * EMB * 2);
  bf16_t* Wupb   = (bf16_t*)walloc((size_t)EMB * LAT * 2);
  // intermediates
  bf16_t* bufG   = (bf16_t*)walloc((size_t)TOKENS * HID * 2);   // sqrelu(gate) bf16
  bf16_t* bufH   = (bf16_t*)walloc((size_t)TOKENS * HID * 2);   // combined hidden bf16
  bf16_t* xl     = (bf16_t*)walloc((size_t)TOKENS * LAT * 2);   // latent tokens bf16
  float*  accL   = (float*) walloc((size_t)TOKENS * LAT * 4);   // routed accumulator f32
  bf16_t* accLb  = (bf16_t*)walloc((size_t)TOKENS * LAT * 2);
  float*  probas = (float*) walloc((size_t)TOKENS * NEXP * 4);
  float*  wdense = (float*) walloc((size_t)TOKENS * NEXP * 4);

  dim3 blk(256);
  auto grid = [](int M, int N) { return dim3(N / TNB, M / TMB); };
  auto cvt = [&](const float* s, bf16_t* d, long n) {
    cvt_bf16_kernel<<<(n / 8 + 255) / 256, 256, 0, stream>>>(s, d, n);
  };

  // one-shot bf16 conversions (weights are reused 32x per GEMM; cost ~15us of BW)
  cvt(x, xbf, (long)TOKENS * EMB);
  cvt(sw1, sw1b, (long)HID * EMB);
  cvt(swg, swgb, (long)HID * EMB);
  cvt(sw2, sw2b, (long)EMB * HID);
  cvt(rw1, rw1b, (long)NEXP * HID * LAT);
  cvt(rwg, rwgb, (long)NEXP * HID * LAT);
  cvt(rw2, rw2b, (long)NEXP * LAT * HID);
  cvt(Wdown, Wdownb, (long)LAT * EMB);
  cvt(Wup, Wupb, (long)EMB * LAT);

  // gate + routing + accumulator init (fp32 for exact top-k behavior)
  gate_kernel<<<TOKENS, 128, 0, stream>>>(x, Wg, probas);
  route_kernel<<<TOKENS / 256, 256, 0, stream>>>(probas, biases, wdense);
  fill0_kernel<<<((long)TOKENS * LAT + 255) / 256, 256, 0, stream>>>(accL, (long)TOKENS * LAT);

  // shared expert: hidden = (x @ sw1^T) * sqrelu(x @ swg^T); out = hidden @ sw2^T
  gemm_bf16_wmma<EPI_SQRELU><<<grid(TOKENS, HID), blk, 0, stream>>>(
      xbf, swgb, bufG, nullptr, nullptr, 0, TOKENS, HID, EMB);
  gemm_bf16_wmma<EPI_MUL><<<grid(TOKENS, HID), blk, 0, stream>>>(
      xbf, sw1b, bufH, bufG, nullptr, 0, TOKENS, HID, EMB);
  gemm_bf16_wmma<EPI_F32><<<grid(TOKENS, EMB), blk, 0, stream>>>(
      bufH, sw2b, out, nullptr, nullptr, 0, TOKENS, EMB, HID);

  // latent down-projection: xl = x @ Wdown^T (bf16)
  gemm_bf16_wmma<EPI_BF16><<<grid(TOKENS, LAT), blk, 0, stream>>>(
      xbf, Wdownb, xl, nullptr, nullptr, 0, TOKENS, LAT, EMB);

  // routed experts (dense over all 16; combine weight folded into down-proj epilogue)
  for (int e = 0; e < NEXP; e++) {
    const bf16_t* w1e = rw1b + (size_t)e * HID * LAT;
    const bf16_t* wge = rwgb + (size_t)e * HID * LAT;
    const bf16_t* w2e = rw2b + (size_t)e * LAT * HID;
    gemm_bf16_wmma<EPI_SQRELU><<<grid(TOKENS, HID), blk, 0, stream>>>(
        xl, wge, bufG, nullptr, nullptr, 0, TOKENS, HID, LAT);
    gemm_bf16_wmma<EPI_MUL><<<grid(TOKENS, HID), blk, 0, stream>>>(
        xl, w1e, bufH, bufG, nullptr, 0, TOKENS, HID, LAT);
    gemm_bf16_wmma<EPI_ADDRS><<<grid(TOKENS, LAT), blk, 0, stream>>>(
        bufH, w2e, accL, nullptr, wdense + e, NEXP, TOKENS, LAT, HID);
  }

  // out += accL @ Wup^T
  cvt(accL, accLb, (long)TOKENS * LAT);
  gemm_bf16_wmma<EPI_ADDF32><<<grid(TOKENS, EMB), blk, 0, stream>>>(
      accLb, Wupb, out, nullptr, nullptr, 0, TOKENS, EMB, LAT);
}
